// SSMoE_Core_38062000177277
// MI455X (gfx1250) — compile-verified, matrix operand order checked by reading, not run on hardware
//
#include <hip/hip_runtime.h>
#include <hip/hip_bf16.h>

typedef __attribute__((ext_vector_type(16))) __bf16 v16bf;
typedef __attribute__((ext_vector_type(8)))  __bf16 v8bf;
typedef __attribute__((ext_vector_type(4)))  __bf16 v4bf;
typedef __attribute__((ext_vector_type(8)))  float  v8f;
typedef __attribute__((ext_vector_type(4)))  int    v4i;

// AS-qualified pointee typedefs: attribute sticks when applied via typedef.
typedef __attribute__((address_space(1))) v4i as1_v4i;
typedef __attribute__((address_space(3))) v4i as3_v4i;

#define N_TOK   16384
#define DIM     512
#define E_SPEC  8
#define E_SHARED 2
#define E_TOT   10

#define TM 128   // output tile rows per workgroup
#define TN 64    // output tile cols per workgroup
#define TK 32    // K step (bf16 WMMA K)
#define AS_STRIDE 56  // bf16 elems/row: 112B = 7x16B (aligned, conflict-free frag loads)
#define BS_STRIDE 56
#define NSTAGE (E_TOT * (DIM / TK))  // 160 pipeline stages

// ---- CDNA5 async global->LDS path (ASYNCcnt), with safe fallback ----------
#if __has_builtin(__builtin_amdgcn_global_load_async_to_lds_b128)
#define USE_ASYNC_LDS 1
#else
#define USE_ASYNC_LDS 0
#endif

__device__ __forceinline__ void async_ld16(const __bf16* src, __bf16* dst) {
#if USE_ASYNC_LDS
  __builtin_amdgcn_global_load_async_to_lds_b128(
      (as1_v4i*)src, (as3_v4i*)dst, 0, 0);
#else
  *(uint4*)dst = *(const uint4*)src;
#endif
}

template <int N>
__device__ __forceinline__ void wait_async() {
#if USE_ASYNC_LDS
#if __has_builtin(__builtin_amdgcn_s_wait_asynccnt)
  __builtin_amdgcn_s_wait_asynccnt((unsigned short)N);
#else
  asm volatile("s_wait_asynccnt %0" ::"i"(N) : "memory");
#endif
#endif
}

// ---------------------------------------------------------------------------
// Kernel 1: routing. spec softmax -> top2 -> renorm; shared softmax.
// Writes dense weight table wts[N][16] (0..7 spec, 8..9 shared, rest zero).
// ---------------------------------------------------------------------------
__global__ __launch_bounds__(256) void moe_route(
    const float* __restrict__ spec_logits,
    const float* __restrict__ shared_logits,
    float* __restrict__ wts) {
  int n = blockIdx.x * blockDim.x + threadIdx.x;
  if (n >= N_TOK) return;

  float p[E_SPEC];
  float mx = -1e30f;
#pragma unroll
  for (int e = 0; e < E_SPEC; ++e) { p[e] = spec_logits[n * E_SPEC + e]; mx = fmaxf(mx, p[e]); }
  float s = 0.f;
#pragma unroll
  for (int e = 0; e < E_SPEC; ++e) { p[e] = __expf(p[e] - mx); s += p[e]; }
  float inv = 1.f / s;
  int i1 = 0; float v1 = -1.f;
#pragma unroll
  for (int e = 0; e < E_SPEC; ++e) { p[e] *= inv; if (p[e] > v1) { v1 = p[e]; i1 = e; } }
  int i2 = 0; float v2 = -1.f;
#pragma unroll
  for (int e = 0; e < E_SPEC; ++e) { if (e != i1 && p[e] > v2) { v2 = p[e]; i2 = e; } }
  float rn = 1.f / (v1 + v2 + 1e-6f);

  float w[16];
#pragma unroll
  for (int e = 0; e < 16; ++e) w[e] = 0.f;
  w[i1] = v1 * rn;
  w[i2] = v2 * rn;

  float q0 = shared_logits[n * 2 + 0], q1 = shared_logits[n * 2 + 1];
  float m2 = fmaxf(q0, q1);
  float e0 = __expf(q0 - m2), e1 = __expf(q1 - m2);
  float si = 1.f / (e0 + e1);
  w[E_SPEC + 0] = e0 * si;
  w[E_SPEC + 1] = e1 * si;

#pragma unroll
  for (int e = 0; e < 16; ++e) wts[n * 16 + e] = w[e];
}

// ---------------------------------------------------------------------------
// Kernel 2: x f32 -> bf16 (vectorized 4-wide).
// ---------------------------------------------------------------------------
__global__ __launch_bounds__(256) void cvt_x(const float* __restrict__ x,
                                             __bf16* __restrict__ xb) {
  size_t i = ((size_t)blockIdx.x * blockDim.x + threadIdx.x) * 4;
  float4 v = *(const float4*)(x + i);
  v4bf o;
  o[0] = (__bf16)v.x; o[1] = (__bf16)v.y; o[2] = (__bf16)v.z; o[3] = (__bf16)v.w;
  *(v4bf*)(xb + i) = o;
}

// ---------------------------------------------------------------------------
// Kernel 3: transpose + convert expert weights.
// W[e][d][f] (f32) -> Wt[e][f][d] (bf16), LDS-tiled for coalescing.
// grid: (DIM/32, DIM/32, E_TOT), block (32,8)
// ---------------------------------------------------------------------------
__global__ __launch_bounds__(256) void cvt_w(const float* __restrict__ specW,
                                             const float* __restrict__ sharedW,
                                             __bf16* __restrict__ Wt) {
  __shared__ float tile[32][33];
  int e = blockIdx.z;
  const float* W = (e < E_SPEC) ? specW + (size_t)e * DIM * DIM
                                : sharedW + (size_t)(e - E_SPEC) * DIM * DIM;
  int f0 = blockIdx.x * 32;
  int d0 = blockIdx.y * 32;
  for (int r = threadIdx.y; r < 32; r += 8)
    tile[r][threadIdx.x] = W[(size_t)(d0 + r) * DIM + f0 + threadIdx.x];
  __syncthreads();
  for (int r = threadIdx.y; r < 32; r += 8)
    Wt[(size_t)e * DIM * DIM + (size_t)(f0 + r) * DIM + d0 + threadIdx.x] =
        (__bf16)tile[threadIdx.x][r];
}

// ---------------------------------------------------------------------------
// Kernel 4: fused dense MoE GEMM + weighted combine, software-pipelined with
// async global->LDS double buffering across a flattened (expert, k) loop.
//   out[n,f] = sum_e w[n,e] * ( sum_d x[n,d]*W[e,d,f] + b[e,f] )
// 256 threads = 8 waves; wave grid 4x2; wave computes 32x32 via 2x2 WMMA frags.
// ---------------------------------------------------------------------------
__global__ __launch_bounds__(256) void moe_gemm(
    const __bf16* __restrict__ xb,       // [N][DIM] bf16
    const __bf16* __restrict__ Wt,       // [E_TOT][DIM(f)][DIM(d)] bf16
    const float*  __restrict__ wts,      // [N][16]
    const float*  __restrict__ spec_b,   // [8][DIM]
    const float*  __restrict__ shared_b, // [2][DIM]
    float* __restrict__ out) {           // [N][DIM]
  __shared__ __bf16 As[2][TM * AS_STRIDE];
  __shared__ __bf16 Bs[2][TN * BS_STRIDE];
  __shared__ float  Wls[TM * 16];
  __shared__ float  Bls[E_TOT * TN];

  const int tid  = threadIdx.x;
  const int lane = tid & 31;
  const int wave = tid >> 5;
  const int l    = lane & 15;
  const int h    = lane >> 4;
  const int wm   = wave >> 1;  // 0..3
  const int wn   = wave & 1;   // 0..1
  const int tile_m = blockIdx.y * TM;
  const int tile_n = blockIdx.x * TN;

  // per-thread staging addresses: 3 x b128 per lane per stage
  const int s_row  = tid >> 2;          // 0..63
  const int s_seg  = (tid & 3) * 8;     // element offset within 32-elem row
  const __bf16* a_src0 = xb + (size_t)(tile_m + s_row) * DIM + s_seg;
  const __bf16* a_src1 = xb + (size_t)(tile_m + s_row + 64) * DIM + s_seg;
  const __bf16* b_src  = Wt + (size_t)(tile_n + s_row) * DIM + s_seg;

  auto issue = [&](int s, int buf) {
    const int e  = s >> 4;
    const int k0 = (s & 15) * TK;
    async_ld16(a_src0 + k0, &As[buf][s_row * AS_STRIDE + s_seg]);
    async_ld16(a_src1 + k0, &As[buf][(s_row + 64) * AS_STRIDE + s_seg]);
    async_ld16(b_src + (size_t)e * DIM * DIM + k0,
               &Bs[buf][s_row * BS_STRIDE + s_seg]);
  };

  // cache routing weights (128 rows x 16) and biases (10 x 64 cols) in LDS
  for (int i = tid; i < TM * 16; i += 256)
    Wls[i] = wts[(size_t)(tile_m + (i >> 4)) * 16 + (i & 15)];
  for (int i = tid; i < E_TOT * TN; i += 256) {
    int e = i / TN, c = i - e * TN;
    const float* bp = (e < E_SPEC) ? (spec_b + (size_t)e * DIM)
                                   : (shared_b + (size_t)(e - E_SPEC) * DIM);
    Bls[i] = bp[tile_n + c];
  }

  issue(0, 0);  // prologue

  const v8f vzero = {};
  float outacc[2][2][8] = {};
  v8f acc[2][2] = {};

  for (int s = 0; s < NSTAGE; ++s) {
    const int buf = s & 1;
    if (s + 1 < NSTAGE) {
      issue(s + 1, buf ^ 1);
      wait_async<3>();   // stage s complete (per-wave in-order async completion)
    } else {
      wait_async<0>();
    }
    __syncthreads();     // stage-s tiles visible to all waves

    // A fragments: lane (l,h) holds row l, K in [8h,8h+8) U [8h+16,8h+24)
    v16bf af[2], bf[2];
#pragma unroll
    for (int i2 = 0; i2 < 2; ++i2) {
      const __bf16* ap = &As[buf][(wm * 32 + i2 * 16 + l) * AS_STRIDE + h * 8];
      v8bf lo = *(const v8bf*)ap;
      v8bf hi = *(const v8bf*)(ap + 16);
      af[i2] = __builtin_shufflevector(lo, hi, 0, 1, 2, 3, 4, 5, 6, 7,
                                       8, 9, 10, 11, 12, 13, 14, 15);
    }
#pragma unroll
    for (int j = 0; j < 2; ++j) {
      const __bf16* bp = &Bs[buf][(wn * 32 + j * 16 + l) * BS_STRIDE + h * 8];
      v8bf lo = *(const v8bf*)bp;
      v8bf hi = *(const v8bf*)(bp + 16);
      bf[j] = __builtin_shufflevector(lo, hi, 0, 1, 2, 3, 4, 5, 6, 7,
                                      8, 9, 10, 11, 12, 13, 14, 15);
    }
#pragma unroll
    for (int i2 = 0; i2 < 2; ++i2)
#pragma unroll
      for (int j = 0; j < 2; ++j)
        acc[i2][j] = __builtin_amdgcn_wmma_f32_16x16x32_bf16(
            false, af[i2], false, bf[j], (short)0, acc[i2][j], false, false);

    __syncthreads();     // all waves done reading buf before it is overwritten

    if ((s & 15) == 15) {  // finished one expert: weighted combine + reset
      const int e = s >> 4;
#pragma unroll
      for (int i2 = 0; i2 < 2; ++i2)
#pragma unroll
        for (int j = 0; j < 2; ++j) {
          int ncol = wn * 32 + j * 16 + l;
          float bias = Bls[e * TN + ncol];
#pragma unroll
          for (int r = 0; r < 8; ++r) {
            int mrow = wm * 32 + i2 * 16 + r + 8 * h;
            float w = Wls[mrow * 16 + e];
            outacc[i2][j][r] += w * (acc[i2][j][r] + bias);
          }
          acc[i2][j] = vzero;
        }
    }
  }

  // store: C/D layout -> (m = r + 8h, n = l) per 16x16 frag
#pragma unroll
  for (int i2 = 0; i2 < 2; ++i2)
#pragma unroll
    for (int j = 0; j < 2; ++j)
#pragma unroll
      for (int r = 0; r < 8; ++r) {
        int mrow = tile_m + wm * 32 + i2 * 16 + r + 8 * h;
        int ncol = tile_n + wn * 32 + j * 16 + l;
        out[(size_t)mrow * DIM + ncol] = outacc[i2][j][r];
      }
}

// ---------------------------------------------------------------------------
extern "C" void kernel_launch(void* const* d_in, const int* in_sizes, int n_in,
                              void* d_out, int out_size, void* d_ws, size_t ws_size,
                              hipStream_t stream) {
  (void)in_sizes; (void)n_in; (void)out_size; (void)ws_size;

  const float* x             = (const float*)d_in[0];
  const float* spec_logits   = (const float*)d_in[1];
  const float* shared_logits = (const float*)d_in[2];
  const float* spec_W        = (const float*)d_in[3];
  const float* spec_b        = (const float*)d_in[4];
  const float* shared_W      = (const float*)d_in[5];
  const float* shared_b      = (const float*)d_in[6];
  // d_in[7] = top_k (assumed 2, matching the reference setup)

  char* ws = (char*)d_ws;
  __bf16* xb  = (__bf16*)ws;                                     // N*D bf16     (16 MB)
  __bf16* Wtp = (__bf16*)(ws + (size_t)N_TOK * DIM * 2);         // 10*D*D bf16  ( 5 MB)
  float*  wts = (float*)(ws + (size_t)N_TOK * DIM * 2
                            + (size_t)E_TOT * DIM * DIM * 2);    // N*16 f32     ( 1 MB)

  hipLaunchKernelGGL(moe_route, dim3(N_TOK / 256), dim3(256), 0, stream,
                     spec_logits, shared_logits, wts);
  hipLaunchKernelGGL(cvt_x, dim3((N_TOK * DIM / 4) / 256), dim3(256), 0, stream, x, xb);
  hipLaunchKernelGGL(cvt_w, dim3(DIM / 32, DIM / 32, E_TOT), dim3(32, 8), 0, stream,
                     spec_W, shared_W, Wtp);
  hipLaunchKernelGGL(moe_gemm, dim3(DIM / TN, N_TOK / TM), dim3(256), 0, stream,
                     xb, Wtp, wts, spec_b, shared_b, (float*)d_out);
}
